// MSA_76802605187281
// MI455X (gfx1250) — compile-verified, matrix operand order checked by reading, not run on hardware
//
#include <hip/hip_runtime.h>

#define S_LEN 2048
#define NH 16
#define DH 64
#define BATCH 4

#if __has_builtin(__builtin_amdgcn_global_load_async_to_lds_b128) && \
    __has_builtin(__builtin_amdgcn_s_wait_asynccnt)
#define HAVE_ASYNC_LDS 1
#endif

typedef __attribute__((ext_vector_type(16))) __bf16 v16bf;
typedef __attribute__((ext_vector_type(8)))  float  v8f;
typedef __attribute__((ext_vector_type(4)))  int    v4i;

#ifdef HAVE_ASYNC_LDS
typedef __attribute__((address_space(1))) v4i gv4i;   // global-space b128
typedef __attribute__((address_space(3))) v4i lv4i;   // LDS-space b128
#endif

union Frag {
    v16bf v;
    unsigned short u[16];
    unsigned int w[8];
    uint4 q[2];
};

// f32 pair -> packed bf16 pair (one VGPR). Prefer the HW packed convert.
__device__ __forceinline__ unsigned int pk_bf16(float lo, float hi) {
#if __has_builtin(__builtin_amdgcn_cvt_pk_bf16_f32)
    auto r = __builtin_amdgcn_cvt_pk_bf16_f32(lo, hi);
    unsigned int out;
    __builtin_memcpy(&out, &r, sizeof(out));
    return out;
#else
    union { __bf16 b[2]; unsigned int i; } u;
    u.b[0] = (__bf16)lo; u.b[1] = (__bf16)hi;
    return u.i;
#endif
}

__device__ __forceinline__ unsigned short f2bf(float f) {
#if __has_builtin(__builtin_amdgcn_cvt_pk_bf16_f32)
    return (unsigned short)(pk_bf16(f, 0.0f) & 0xFFFFu);
#else
    union { float f; unsigned int i; } a; a.f = f;
    unsigned int r = a.i + 0x7FFFu + ((a.i >> 16) & 1u);
    return (unsigned short)(r >> 16);
#endif
}

__device__ __forceinline__ void pack8(Frag& fr, int w0, float4 a, float4 b) {
    fr.w[w0 + 0] = pk_bf16(a.x, a.y);
    fr.w[w0 + 1] = pk_bf16(a.z, a.w);
    fr.w[w0 + 2] = pk_bf16(b.x, b.y);
    fr.w[w0 + 3] = pk_bf16(b.z, b.w);
}

__device__ __forceinline__ v8f wmma_bf16(const Frag& a, const Frag& b, v8f c) {
    // D = A(16x32 bf16) x B(32x16 bf16) + C(16x16 f32)
    return __builtin_amdgcn_wmma_f32_16x16x32_bf16(
        false, a.v, false, b.v, (short)0, c, false, false);
}

// ---------------------------------------------------------------------------
// Kernel 1: per-head QKV projection (y = x @ W^T + b), fp32 in -> bf16 out.
// Q, K stored [B,H,S,64] row-major; V stored transposed [B,H,64,S].
// One wave per 16-row tile; 8 waves per block.
// ---------------------------------------------------------------------------
__global__ void __launch_bounds__(256)
qkv_proj_kernel(const float* __restrict__ seq,
                const float* __restrict__ Wq, const float* __restrict__ bq,
                const float* __restrict__ Wk, const float* __restrict__ bk,
                const float* __restrict__ Wv, const float* __restrict__ bv,
                unsigned short* __restrict__ Qo,
                unsigned short* __restrict__ Ko,
                unsigned short* __restrict__ VTo) {
    const int lane = threadIdx.x & 31;
    const int wave = threadIdx.x >> 5;
    const int tile = blockIdx.x * 8 + wave;     // 8192 tiles total
    const int s0 = (tile & 127) << 4;           // 128 row-tiles per (b,h)
    const int h  = (tile >> 7) & 15;
    const int b  = tile >> 11;
    const int hf = lane >> 4;
    const int lm = lane & 15;

    // A fragments of X tile: rows s0+lm, contraction d split into 2 chunks of 32
    Frag xa[2];
    {
        const float* xrow = seq + (size_t)(b * S_LEN + (s0 + lm)) * (NH * DH) + h * DH;
        #pragma unroll
        for (int c = 0; c < 2; ++c) {
            float4 f0 = *(const float4*)(xrow + 32 * c + 8 * hf);
            float4 f1 = *(const float4*)(xrow + 32 * c + 8 * hf + 4);
            float4 f2 = *(const float4*)(xrow + 32 * c + 16 + 8 * hf);
            float4 f3 = *(const float4*)(xrow + 32 * c + 16 + 8 * hf + 4);
            pack8(xa[c], 0, f0, f1);
            pack8(xa[c], 4, f2, f3);
        }
    }

    const float* Ws[3] = { Wq, Wk, Wv };
    const float* Bs[3] = { bq, bk, bv };
    unsigned short* Ds[3] = { Qo, Ko, VTo };

    #pragma unroll
    for (int mtx = 0; mtx < 3; ++mtx) {
        const float* W = Ws[mtx];
        const float* bias = Bs[mtx];
        unsigned short* dst = Ds[mtx];
        const bool tr = (mtx == 2);
        #pragma unroll
        for (int et = 0; et < 4; ++et) {
            const int e = et * 16 + lm;                        // B-frag col / C col
            const float* wrow = W + ((size_t)h * DH + e) * DH; // W[h][e][:]
            const float bval = bias[h * DH + e];
            v8f acc;
            #pragma unroll
            for (int j = 0; j < 8; ++j) acc[j] = bval;         // bias preloaded into C
            #pragma unroll
            for (int c = 0; c < 2; ++c) {
                Frag bf;
                float4 f0 = *(const float4*)(wrow + 32 * c + 8 * hf);
                float4 f1 = *(const float4*)(wrow + 32 * c + 8 * hf + 4);
                float4 f2 = *(const float4*)(wrow + 32 * c + 16 + 8 * hf);
                float4 f3 = *(const float4*)(wrow + 32 * c + 16 + 8 * hf + 4);
                pack8(bf, 0, f0, f1);
                pack8(bf, 4, f2, f3);
                acc = wmma_bf16(xa[c], bf, acc);
            }
            // C layout: lane -> col n=lm, VGPR j -> row m = j + 8*hf
            #pragma unroll
            for (int j = 0; j < 8; ++j) {
                const int row = s0 + j + 8 * hf;
                const size_t idx = tr
                    ? ((size_t)((b * NH + h) * DH + e) * S_LEN + row)
                    : ((size_t)((b * NH + h) * S_LEN + row) * DH + e);
                dst[idx] = f2bf(acc[j]);
            }
        }
    }
}

// ---------------------------------------------------------------------------
// Kernel 2: fused flash attention. 4 waves per block, all on consecutive
// 16-query tiles of the SAME (b,h): K/V chunks staged into LDS once per block
// via ASYNC global->LDS DMA when available (no VGPR round-trip, ASYNCcnt),
// else via regular load+ds_store. Fragments built with ds_load_b128.
// ---------------------------------------------------------------------------
__global__ void __launch_bounds__(128)
flash_attn_kernel(const unsigned short* __restrict__ Q,
                  const unsigned short* __restrict__ K,
                  const unsigned short* __restrict__ VT,
                  float* __restrict__ out) {
    __shared__ unsigned short kbuf[32 * 64];        // K chunk  [key][d]   4KB
    __shared__ unsigned short vbuf[64 * 32];        // V chunk  [d][key]   4KB
    __shared__ unsigned short pbuf[4][16 * 32];     // per-wave P transpose 4KB

    const int tid  = threadIdx.x;
    const int lane = tid & 31;
    const int wave = tid >> 5;
    const int tile = blockIdx.x * 4 + wave;         // 8192 tiles; 4/block share (b,h)
    const int s0 = (tile & 127) << 4;
    const int h  = (tile >> 7) & 15;
    const int b  = tile >> 11;
    const int hf = lane >> 4;
    const int lm = lane & 15;

    const unsigned short* Qb = Q  + (size_t)(b * NH + h) * S_LEN * DH;
    const unsigned short* Kb = K  + (size_t)(b * NH + h) * S_LEN * DH;
    const unsigned short* Vb = VT + (size_t)(b * NH + h) * DH * S_LEN;

    // Q A-fragments, resident for the whole key loop
    Frag qa[2];
    {
        const unsigned short* qrow = Qb + (size_t)(s0 + lm) * DH;
        #pragma unroll
        for (int c = 0; c < 2; ++c) {
            qa[c].q[0] = *(const uint4*)(qrow + 32 * c + 8 * hf);
            qa[c].q[1] = *(const uint4*)(qrow + 32 * c + 16 + 8 * hf);
        }
    }

    v8f o[4];
    v8f mrow, lrow;
    #pragma unroll
    for (int et = 0; et < 4; ++et)
        #pragma unroll
        for (int j = 0; j < 8; ++j) o[et][j] = 0.0f;
    #pragma unroll
    for (int j = 0; j < 8; ++j) { mrow[j] = -3.0e38f; lrow[j] = 0.0f; }

    for (int kb = 0; kb < S_LEN; kb += 32) {
        // ---- cooperative staging: K chunk (32x64) + V chunk (64x32) -> LDS ----
#ifdef HAVE_ASYNC_LDS
        #pragma unroll
        for (int r = 0; r < 2; ++r) {
            const int idx = tid + 128 * r;          // 0..255 b128 slots
            const int key = idx >> 3, seg = idx & 7;
            __builtin_amdgcn_global_load_async_to_lds_b128(
                (gv4i*)(Kb + (size_t)(kb + key) * DH + seg * 8),
                (lv4i*)(kbuf + key * 64 + seg * 8),
                0, 0);
        }
        #pragma unroll
        for (int r = 0; r < 2; ++r) {
            const int idx = tid + 128 * r;
            const int d = idx >> 2, seg = idx & 3;
            __builtin_amdgcn_global_load_async_to_lds_b128(
                (gv4i*)(Vb + (size_t)d * S_LEN + kb + seg * 8),
                (lv4i*)(vbuf + d * 32 + seg * 8),
                0, 0);
        }
        __builtin_amdgcn_s_wait_asynccnt(0);
#else
        #pragma unroll
        for (int r = 0; r < 2; ++r) {
            const int idx = tid + 128 * r;          // 0..255 b128 slots
            const int key = idx >> 3, seg = idx & 7;
            *(uint4*)(kbuf + key * 64 + seg * 8) =
                *(const uint4*)(Kb + (size_t)(kb + key) * DH + seg * 8);
        }
        #pragma unroll
        for (int r = 0; r < 2; ++r) {
            const int idx = tid + 128 * r;
            const int d = idx >> 2, seg = idx & 3;
            *(uint4*)(vbuf + d * 32 + seg * 8) =
                *(const uint4*)(Vb + (size_t)d * S_LEN + kb + seg * 8);
        }
#endif
        if (kb + 32 < S_LEN) {                      // prefetch next chunk into cache
            __builtin_prefetch(Kb + (size_t)(kb + 32) * DH + tid * 16, 0, 1);
            __builtin_prefetch(Vb + (size_t)(tid >> 1) * S_LEN + kb + 32 + (tid & 1) * 16, 0, 1);
        }
        __syncthreads();

        // ---- scores: two 16x16 tiles from LDS-resident K ----
        v8f sc[2];
        #pragma unroll
        for (int t = 0; t < 2; ++t) {
            v8f acc;
            #pragma unroll
            for (int j = 0; j < 8; ++j) acc[j] = 0.0f;
            const unsigned short* krow = kbuf + (16 * t + lm) * 64;
            #pragma unroll
            for (int c = 0; c < 2; ++c) {
                Frag kf;
                kf.q[0] = *(const uint4*)(krow + 32 * c + 8 * hf);
                kf.q[1] = *(const uint4*)(krow + 32 * c + 16 + 8 * hf);
                acc = wmma_bf16(qa[c], kf, acc);
            }
            #pragma unroll
            for (int j = 0; j < 8; ++j) sc[t][j] = acc[j] * 0.125f;  // 1/sqrt(64)
        }

        // ---- online softmax (reductions within 16-lane groups = C-layout rows) ----
        v8f cmax;
        #pragma unroll
        for (int j = 0; j < 8; ++j) cmax[j] = fmaxf(sc[0][j], sc[1][j]);
        #pragma unroll
        for (int msk = 1; msk < 16; msk <<= 1)
            #pragma unroll
            for (int j = 0; j < 8; ++j)
                cmax[j] = fmaxf(cmax[j], __shfl_xor(cmax[j], msk, 32));

        v8f mnew, alpha, p0, p1, rsum;
        #pragma unroll
        for (int j = 0; j < 8; ++j) {
            mnew[j]  = fmaxf(mrow[j], cmax[j]);
            alpha[j] = __expf(mrow[j] - mnew[j]);
            p0[j]    = __expf(sc[0][j] - mnew[j]);
            p1[j]    = __expf(sc[1][j] - mnew[j]);
            rsum[j]  = p0[j] + p1[j];
        }
        #pragma unroll
        for (int msk = 1; msk < 16; msk <<= 1)
            #pragma unroll
            for (int j = 0; j < 8; ++j)
                rsum[j] += __shfl_xor(rsum[j], msk, 32);
        #pragma unroll
        for (int j = 0; j < 8; ++j) {
            lrow[j] = lrow[j] * alpha[j] + rsum[j];
            mrow[j] = mnew[j];
        }
        #pragma unroll
        for (int et = 0; et < 4; ++et)
            #pragma unroll
            for (int j = 0; j < 8; ++j) o[et][j] *= alpha[j];

        // ---- transpose P (C-layout) -> A-layout via per-wave LDS bounce ----
        unsigned short* pb = pbuf[wave];
        #pragma unroll
        for (int j = 0; j < 8; ++j) {
            const int m = j + 8 * hf;
            pb[m * 32 + lm]      = f2bf(p0[j]);
            pb[m * 32 + 16 + lm] = f2bf(p1[j]);
        }
        __builtin_amdgcn_wave_barrier();    // same-wave DS ops are in-order
        Frag pf;
        pf.q[0] = *(const uint4*)(pb + lm * 32 + 8 * hf);
        pf.q[1] = *(const uint4*)(pb + lm * 32 + 16 + 8 * hf);

        // ---- O += P(16x32) x V(32x64) from LDS-resident V^T ----
        #pragma unroll
        for (int et = 0; et < 4; ++et) {
            Frag vf;
            const unsigned short* vrow = vbuf + (et * 16 + lm) * 32;
            vf.q[0] = *(const uint4*)(vrow + 8 * hf);
            vf.q[1] = *(const uint4*)(vrow + 16 + 8 * hf);
            o[et] = wmma_bf16(pf, vf, o[et]);
        }
        __syncthreads();                    // protect K/V buffers before restage
    }

    // ---- epilogue: normalize and write fp32 output [B,S,HIDDEN] ----
    v8f inv;
    #pragma unroll
    for (int j = 0; j < 8; ++j) inv[j] = 1.0f / lrow[j];
    #pragma unroll
    for (int et = 0; et < 4; ++et)
        #pragma unroll
        for (int j = 0; j < 8; ++j) {
            const int row = s0 + j + 8 * hf;
            out[(size_t)(b * S_LEN + row) * (NH * DH) + h * DH + et * 16 + lm]
                = o[et][j] * inv[j];
        }
}

extern "C" void kernel_launch(void* const* d_in, const int* in_sizes, int n_in,
                              void* d_out, int out_size, void* d_ws, size_t ws_size,
                              hipStream_t stream) {
    (void)in_sizes; (void)n_in; (void)out_size; (void)ws_size;
    const float* seq = (const float*)d_in[0];
    const float* Wq  = (const float*)d_in[1];
    const float* bq  = (const float*)d_in[2];
    const float* Wk  = (const float*)d_in[3];
    const float* bk  = (const float*)d_in[4];
    const float* Wv  = (const float*)d_in[5];
    const float* bv  = (const float*)d_in[6];
    float* out = (float*)d_out;

    const size_t per = (size_t)BATCH * NH * S_LEN * DH;   // elements per Q/K/V buffer
    unsigned short* Qw = (unsigned short*)d_ws;
    unsigned short* Kw = Qw + per;
    unsigned short* Vw = Kw + per;

    qkv_proj_kernel<<<1024, 256, 0, stream>>>(seq, Wq, bq, Wk, bk, Wv, bv, Qw, Kw, Vw);
    flash_attn_kernel<<<2048, 128, 0, stream>>>(Qw, Kw, Vw, out);
}
